// CombinedModel_19731079758336
// MI455X (gfx1250) — compile-verified, matrix operand order checked by reading, not run on hardware
//
#include <hip/hip_runtime.h>
#include <hip/hip_bf16.h>

typedef __bf16 bf16;
typedef __attribute__((ext_vector_type(16))) __bf16 v16bf;
typedef __attribute__((ext_vector_type(8)))  float  v8f;

#define N_BOX   2048
#define S_TOT   (N_BOX * 7)        // 14336
#define FLATN   512
#define HDIM    128
#define NHEAD   8
#define OUTC    273                // 38 + 25 + 6*35
#define IMG_W   640
#define IMG_CH_STRIDE (640 * 640)

#define KSLICE  128                // K elements staged to LDS per phase
#define BROW    136                // LDS row stride in bf16 elems (272B: bank-conflict-free)

// ---------------------------------------------------------------------------
// Kernel 1: detection loss + confidence mask (deterministic single block)
// ---------------------------------------------------------------------------
__global__ __launch_bounds__(256) void loss_kernel(
    const int* __restrict__ box_x, const int* __restrict__ box_y,
    const float* __restrict__ targets, int* __restrict__ mask,
    float* __restrict__ out_loss) {
  __shared__ float red[256];
  const int tid = threadIdx.x;
  float local = 0.f;
  for (int n = tid; n < N_BOX; n += 256) {
    const float fx = (float)box_x[n];
    const float fy = (float)box_y[n];
    const float* t = targets + n * 4;
    const float d0 = fx - t[0];
    const float d1 = fy - t[1];
    const float d2 = (fx + 128.f) - t[2];
    const float d3 = (fy + 224.f) - t[3];
    const float li = 0.25f * (d0 * d0 + d1 * d1 + d2 * d2 + d3 * d3);
    mask[n] = (li > 0.25f) ? 1 : 0;
    local += li;
  }
  red[tid] = local;
  __syncthreads();
  for (int off = 128; off > 0; off >>= 1) {
    if (tid < off) red[tid] += red[tid + off];
    __syncthreads();
  }
  if (tid == 0) *out_loss = red[0];
}

// ---------------------------------------------------------------------------
// Kernel 2: W1 [8][512][128] f32 -> W1t [8][128][512] bf16 (K contiguous)
// ---------------------------------------------------------------------------
__global__ __launch_bounds__(256) void w1t_kernel(
    const float* __restrict__ W1, bf16* __restrict__ w1t) {
  const int idx = blockIdx.x * 256 + threadIdx.x;   // 8*128*512 = 524288 total
  const int k = idx >> 16;          // head
  const int rem = idx & 65535;
  const int n = rem >> 9;           // hidden col
  const int f = rem & 511;          // feature (K)
  w1t[idx] = (bf16)W1[((size_t)k * FLATN + f) * HDIM + n];
}

// ---------------------------------------------------------------------------
// Kernel 3: per strip: adaptive maxpool(32x128->8x16) + conv3x3(3->16) + relu
//           + maxpool2 -> flat[512] in bf16.  One block per strip.
// ---------------------------------------------------------------------------
__global__ __launch_bounds__(128) void strip_kernel(
    const float* __restrict__ img, const int* __restrict__ box_x,
    const int* __restrict__ box_y, const float* __restrict__ conv_w,
    const float* __restrict__ conv_b, bf16* __restrict__ flat) {
  const int s  = blockIdx.x;
  const int nb = s / 7;
  const int js = s - nb * 7;
  const int bx = box_x[nb];
  const int by = box_y[nb] + js * 32;
  const int tid = threadIdx.x;

  __shared__ float pool[3][8][16];
  __shared__ float wsm[432];
  __shared__ float bsm[16];

  for (int i = tid; i < 432; i += 128) wsm[i] = conv_w[i];
  if (tid < 16) bsm[tid] = conv_b[tid];

  // adaptive max pool: bins of 4x8
  for (int e = tid; e < 384; e += 128) {
    const int c  = e >> 7;
    const int ph = (e >> 4) & 7;
    const int pw = e & 15;
    const float* base = img + (size_t)c * IMG_CH_STRIDE
                      + (size_t)(by + ph * 4) * IMG_W + (bx + pw * 8);
    float m = -1.0e30f;
#pragma unroll
    for (int dy = 0; dy < 4; ++dy) {
      const float* r = base + dy * IMG_W;
#pragma unroll
      for (int dx = 0; dx < 8; ++dx) m = fmaxf(m, r[dx]);
    }
    pool[c][ph][pw] = m;
  }
  __syncthreads();

  // conv3x3 SAME + bias + relu + maxpool2 -> flat index f = o*32 + py*8 + px
  for (int f = tid; f < FLATN; f += 128) {
    const int o  = f >> 5;
    const int py = (f >> 3) & 3;
    const int px = f & 7;
    float best = -1.0e30f;
#pragma unroll
    for (int dy2 = 0; dy2 < 2; ++dy2) {
#pragma unroll
      for (int dx2 = 0; dx2 < 2; ++dx2) {
        const int y = py * 2 + dy2;
        const int x = px * 2 + dx2;
        float acc = bsm[o];
#pragma unroll
        for (int c = 0; c < 3; ++c) {
#pragma unroll
          for (int ky = 0; ky < 3; ++ky) {
            const int yy = y + ky - 1;
            if (yy < 0 || yy > 7) continue;
#pragma unroll
            for (int kx = 0; kx < 3; ++kx) {
              const int xx = x + kx - 1;
              if (xx < 0 || xx > 15) continue;
              acc += pool[c][yy][xx] * wsm[((o * 3 + c) * 3 + ky) * 3 + kx];
            }
          }
        }
        acc = fmaxf(acc, 0.f);
        best = fmaxf(best, acc);
      }
    }
    flat[(size_t)s * FLATN + f] = (bf16)best;
  }
}

// ---------------------------------------------------------------------------
// Kernel 4: WMMA GEMM  h[k] = relu(flat @ W1[k] + b1[k])
//   Block = 8 waves, one head; each wave owns TWO 16-row S-tiles (32 rows),
//   so every LDS B-fragment feeds 2 WMMAs.  B staged to LDS in 4 K-phases
//   via global_load_async_to_lds_b128; inner K-loop software-pipelined with
//   ping-pong fragment buffers + sched barriers (loads overlap WMMAs).
// ---------------------------------------------------------------------------
__device__ __forceinline__ v16bf load_frag_g(const bf16* p) {
  v16bf r;
  uint4* d = reinterpret_cast<uint4*>(&r);
  d[0] = *reinterpret_cast<const uint4*>(p);        // frag elems 0..7
  d[1] = *reinterpret_cast<const uint4*>(p + 16);   // frag elems 8..15 (K +16)
  return r;
}

__device__ __forceinline__ v16bf load_frag_l(const bf16* p) {
  v16bf r;
  uint4* d = reinterpret_cast<uint4*>(&r);
  d[0] = *reinterpret_cast<const uint4*>(p);
  d[1] = *reinterpret_cast<const uint4*>(p + 16);
  return r;
}

__global__ __launch_bounds__(256) void gemm_w1_kernel(
    const bf16* __restrict__ flat, const bf16* __restrict__ w1t,
    const float* __restrict__ b1, float* __restrict__ h) {
  __shared__ bf16 Bsm[HDIM * BROW];       // 128 rows * 272B = 34,816B

  const int tid   = threadIdx.x;
  const int lane  = tid & 31;
  const int wave  = tid >> 5;             // 0..7
  const int head  = blockIdx.y;
  const int sBase = (blockIdx.x * 8 + wave) * 32;   // 32 rows per wave
  const int rsel  = lane & 15;            // A row / B col selector
  const int khalf = (lane >> 4) * 8;      // lanes 16-31 hold K blocks +8

  // A: ISA 7.12.2 16-bit layout: lane<16 -> K {0..7,16..23}, lane>=16 -> {8..15,24..31}
  const bf16* aRow0 = flat + (size_t)(sBase + rsel) * FLATN + khalf;
  const bf16* aRow1 = aRow0 + (size_t)16 * FLATN;
  const unsigned long long bGlobal =
      (unsigned long long)(w1t + (size_t)head * HDIM * FLATN);

  v8f acc[2][8];
  const v8f zero = {0.f, 0.f, 0.f, 0.f, 0.f, 0.f, 0.f, 0.f};
#pragma unroll
  for (int t = 0; t < 2; ++t)
#pragma unroll
    for (int n = 0; n < 8; ++n) acc[t][n] = zero;

  for (int phase = 0; phase < 4; ++phase) {
    // ---- async-stage 128 rows x 128 K-elems of B into LDS ----
    // 16B chunks: 16 per row, 2048 total, 8 per thread
#pragma unroll
    for (int it = 0; it < 8; ++it) {
      const int chunk = tid + it * 256;
      const int row = chunk >> 4;
      const int cc  = chunk & 15;
      const unsigned lds_addr =
          (unsigned)(unsigned long long)&Bsm[row * BROW + cc * 8];
      const unsigned goff =
          (unsigned)(((unsigned)(row * FLATN) + phase * KSLICE + cc * 8) * 2u);
      asm volatile("global_load_async_to_lds_b128 %0, %1, %2"
                   :: "v"(lds_addr), "v"(goff), "s"(bGlobal) : "memory");
    }
    asm volatile("s_wait_asynccnt 0x0" ::: "memory");
    __syncthreads();

    // ---- software-pipelined 4 K-steps of 32 over this phase ----
    v16bf a0[2], a1[2], bfr[2][8];

    // prologue: load step 0 fragments
    a0[0] = load_frag_g(aRow0 + phase * KSLICE);
    a1[0] = load_frag_g(aRow1 + phase * KSLICE);
#pragma unroll
    for (int n = 0; n < 8; ++n)
      bfr[0][n] = load_frag_l(&Bsm[(n * 16 + rsel) * BROW + khalf]);

#pragma unroll
    for (int step = 0; step < 4; ++step) {
      const int cur = step & 1;
      const int nxt = cur ^ 1;
      if (step < 3) {   // prefetch next K-step while current WMMAs run
        const int kk = (step + 1) * 32;
        a0[nxt] = load_frag_g(aRow0 + phase * KSLICE + kk);
        a1[nxt] = load_frag_g(aRow1 + phase * KSLICE + kk);
#pragma unroll
        for (int n = 0; n < 8; ++n)
          bfr[nxt][n] = load_frag_l(&Bsm[(n * 16 + rsel) * BROW + kk + khalf]);
      }
      __builtin_amdgcn_sched_barrier(0);   // keep prefetch above the WMMA group
#pragma unroll
      for (int n = 0; n < 8; ++n) {
        acc[0][n] = __builtin_amdgcn_wmma_f32_16x16x32_bf16(
            false, a0[cur], false, bfr[cur][n], (short)0, acc[0][n], false, false);
        acc[1][n] = __builtin_amdgcn_wmma_f32_16x16x32_bf16(
            false, a1[cur], false, bfr[cur][n], (short)0, acc[1][n], false, false);
      }
      __builtin_amdgcn_sched_barrier(0);
    }
    __syncthreads();   // before next phase overwrites Bsm
  }

  // C layout: VGPR r -> rows r (lanes 0-15) / r+8 (lanes 16-31), col = lane%16
  const int rowOff = (lane < 16) ? 0 : 8;
#pragma unroll
  for (int t = 0; t < 2; ++t) {
#pragma unroll
    for (int n = 0; n < 8; ++n) {
      const int col = n * 16 + rsel;
      const float bias = b1[head * HDIM + col];
      float* dst = h + ((size_t)head * S_TOT + sBase + t * 16 + rowOff) * HDIM + col;
#pragma unroll
      for (int r = 0; r < 8; ++r) {
        const float v = acc[t][n][r] + bias;
        dst[(size_t)r * HDIM] = v > 0.f ? v : 0.f;
      }
    }
  }
}

// ---------------------------------------------------------------------------
// Kernel 5: head layer + confidence masking -> logits [S,273]
// ---------------------------------------------------------------------------
__global__ __launch_bounds__(288) void heads_kernel(
    const float* __restrict__ h,
    const float* __restrict__ W2_prov,  const float* __restrict__ b2_prov,
    const float* __restrict__ W2_alpha, const float* __restrict__ b2_alpha,
    const float* __restrict__ W2_ad,    const float* __restrict__ b2_ad,
    const int* __restrict__ mask, float* __restrict__ out) {
  const int s   = blockIdx.x;
  const int tid = threadIdx.x;
  float* orow = out + (size_t)s * OUTC;
  if (!mask[s / 7]) {                  // uniform per block
    if (tid < OUTC) orow[tid] = 0.f;
    return;
  }
  __shared__ float hs[NHEAD * HDIM];
  for (int e = tid; e < NHEAD * HDIM; e += 288) {
    const int head = e >> 7;
    const int c = e & 127;
    hs[e] = h[((size_t)head * S_TOT + s) * HDIM + c];
  }
  __syncthreads();

  if (tid < 38) {
    float acc = b2_prov[tid];
    for (int k = 0; k < HDIM; ++k) acc += hs[k] * W2_prov[k * 38 + tid];
    orow[tid] = acc;
  } else if (tid < 63) {
    const int j = tid - 38;
    float acc = b2_alpha[j];
    for (int k = 0; k < HDIM; ++k) acc += hs[HDIM + k] * W2_alpha[k * 25 + j];
    orow[tid] = acc;
  } else if (tid < OUTC) {
    const int r = tid - 63;
    const int t = r / 35;
    const int c = r - t * 35;
    const float* wb = W2_ad + (size_t)t * HDIM * 35 + c;
    const float* hb = hs + (2 + t) * HDIM;
    float acc = b2_ad[t * 35 + c];
    for (int k = 0; k < HDIM; ++k) acc += hb[k] * wb[k * 35];
    orow[tid] = acc;
  }
}

// ---------------------------------------------------------------------------
extern "C" void kernel_launch(void* const* d_in, const int* in_sizes, int n_in,
                              void* d_out, int out_size, void* d_ws, size_t ws_size,
                              hipStream_t stream) {
  const float* x        = (const float*)d_in[0];
  const float* targets  = (const float*)d_in[1];
  const float* conv_w   = (const float*)d_in[2];
  const float* conv_b   = (const float*)d_in[3];
  const float* W1       = (const float*)d_in[4];
  const float* b1       = (const float*)d_in[5];
  const float* W2_prov  = (const float*)d_in[6];
  const float* b2_prov  = (const float*)d_in[7];
  const float* W2_alpha = (const float*)d_in[8];
  const float* b2_alpha = (const float*)d_in[9];
  const float* W2_ad    = (const float*)d_in[10];
  const float* b2_ad    = (const float*)d_in[11];
  const int*   box_x    = (const int*)d_in[12];
  const int*   box_y    = (const int*)d_in[13];
  float* dout = (float*)d_out;

  // workspace layout (all 256B-aligned offsets)
  char* ws = (char*)d_ws;
  bf16*  flatb = (bf16*)ws;                                  // S*512*2   = 14,680,064
  bf16*  w1t   = (bf16*)(ws + 14680064);                     // 8*128*512*2 = 1,048,576
  float* hbuf  = (float*)(ws + 15728640);                    // 8*S*128*4 = 58,720,256
  int*   mask  = (int*)(ws + 74448896);                      // 2048*4

  loss_kernel<<<1, 256, 0, stream>>>(box_x, box_y, targets, mask,
                                     dout + (size_t)S_TOT * OUTC);
  w1t_kernel<<<2048, 256, 0, stream>>>(W1, w1t);
  strip_kernel<<<S_TOT, 128, 0, stream>>>(x, box_x, box_y, conv_w, conv_b, flatb);
  gemm_w1_kernel<<<dim3(S_TOT / 256, NHEAD), 256, 0, stream>>>(flatb, w1t, b1, hbuf);
  heads_kernel<<<S_TOT, 288, 0, stream>>>(hbuf, W2_prov, b2_prov, W2_alpha,
                                          b2_alpha, W2_ad, b2_ad, mask, dout);
}
